// Encoder_80041010528719
// MI455X (gfx1250) — matrix operand order repaired and checked by reading, not compiled, not run on hardware
//
#include <hip/hip_runtime.h>
#include <hip/hip_bf16.h>
#include <math.h>

typedef __bf16 bf16;
typedef bf16  v16bf __attribute__((ext_vector_type(16)));
typedef bf16  v8bf  __attribute__((ext_vector_type(8)));
typedef float v8f   __attribute__((ext_vector_type(8)));

#define TOK   16384
#define EMB_D 1024
#define FF_D  4096
#define NB    16
#define NH    16
#define SQ    1024
#define HD    64

enum { EPI_F32 = 0, EPI_BF16 = 1, EPI_QKV = 2, EPI_QKV_T = 3, EPI_RES = 4, EPI_GELU = 5 };

// ---------------------------------------------------------------------------
// fp32[K,N] -> bf16[N,K] transposed conversion (LDS-tiled, done once per
// weight; everything is L2-resident so this is bandwidth-trivial).
// Makes every WMMA B-fragment a single contiguous 32B vector load.
// ---------------------------------------------------------------------------
__global__ __launch_bounds__(256)
void cvt_transpose_kernel(const float* __restrict__ in, bf16* __restrict__ out,
                          int K, int N) {
    __shared__ bf16 tile[32][33];
    const int kt = blockIdx.y * 32;
    const int nt = blockIdx.x * 32;
    const int tx = threadIdx.x & 31;
    const int ty = threadIdx.x >> 5;   // 0..7
#pragma unroll
    for (int i = 0; i < 4; ++i) {
        int k = kt + ty + 8 * i;
        tile[ty + 8 * i][tx] = (bf16)in[(size_t)k * N + nt + tx];   // coalesced read
    }
    __syncthreads();
#pragma unroll
    for (int i = 0; i < 4; ++i) {
        int n = nt + ty + 8 * i;
        out[(size_t)n * K + kt + tx] = tile[tx][ty + 8 * i];        // coalesced write
    }
}

// ---------------------------------------------------------------------------
// Embedding gather: x_f32 (residual path) + x_bf16 (WMMA operand)
// ---------------------------------------------------------------------------
__global__ void embed_gather_kernel(const int* __restrict__ ids,
                                    const float* __restrict__ table,
                                    float* __restrict__ xf, bf16* __restrict__ xb) {
    int tok = blockIdx.x;
    int id  = ids[tok];
    const float* row = table + (size_t)id * EMB_D;
    for (int e = threadIdx.x; e < EMB_D; e += 256) {
        float v = row[e];
        xf[(size_t)tok * EMB_D + e] = v;
        xb[(size_t)tok * EMB_D + e] = (bf16)v;
    }
}

// ---------------------------------------------------------------------------
// bf16 WMMA GEMM: C[M,N] = A[M,K] @ B[K,N] + bias, with B given TRANSPOSED
// as Bt[N,K]. Block = 256 threads = 8 waves; each wave owns a 16x64 tile
// (4 accumulators; A-fragment reused across the 4 N-tiles).
// Per K=32 step: 2 x 16B A loads + 4 x 32B B loads + 4 WMMAs.
// A-fragment (ISA 16-bit A 16x32): lane = half*16 + m; the lane's 16 bf16
// elements are 16 K-contiguous values (k = 16*half + e), as for B. B-fragment: lane = half*16 + n;
// 16 K-contiguous elements -> one v16bf load from Bt.
// ---------------------------------------------------------------------------
template <int MODE>
__global__ __launch_bounds__(256)
void gemm_bf16_kernel(const bf16* __restrict__ A, const bf16* __restrict__ Bt,
                      const float* __restrict__ bias, int M, int N, int K,
                      void* __restrict__ Cout, const float* __restrict__ resid) {
    const int lane = threadIdx.x & 31;
    const int wave = threadIdx.x >> 5;     // 0..7
    const int half = lane >> 4;
    const int nl   = lane & 15;
    const int r0   = blockIdx.x * 16;
    const int n0   = (blockIdx.y * 8 + wave) * 64;
    if (r0 >= M || n0 >= N) return;

    v8f acc[4] = {};

    for (int k0 = 0; k0 < K; k0 += 32) {
        const bf16* Arow = A + (size_t)(r0 + nl) * K + k0 + 16 * half;
        v8bf alo = *(const v8bf*)(Arow);
        v8bf ahi = *(const v8bf*)(Arow + 8);
        v16bf a;
#pragma unroll
        for (int e = 0; e < 8; ++e) { a[e] = alo[e]; a[8 + e] = ahi[e]; }

#pragma unroll
        for (int t = 0; t < 4; ++t) {
            const bf16* Bp = Bt + (size_t)(n0 + t * 16 + nl) * K + k0 + 16 * half;
            v16bf b = *(const v16bf*)Bp;
            acc[t] = __builtin_amdgcn_wmma_f32_16x16x32_bf16(
                false, a, false, b, (short)0, acc[t], false, false);
        }
    }

    // Epilogue. C layout: VGPR v -> row r0 + v + 8*half, lane nl -> col.
#pragma unroll
    for (int t = 0; t < 4; ++t) {
#pragma unroll
        for (int v = 0; v < 8; ++v) {
            int row = r0 + v + 8 * half;
            int col = n0 + t * 16 + nl;
            float val = acc[t][v] + bias[col];
            size_t idx = (size_t)row * N + col;
            if (MODE == EPI_F32) {
                ((float*)Cout)[idx] = val;
            } else if (MODE == EPI_BF16) {
                ((bf16*)Cout)[idx] = (bf16)val;
            } else if (MODE == EPI_QKV) {
                // [B,H,S,D] (row-major per head)
                int b = row >> 10, s = row & 1023, h = col >> 6, d = col & 63;
                ((bf16*)Cout)[((((size_t)b * NH + h) << 10) + s) * HD + d] = (bf16)val;
            } else if (MODE == EPI_QKV_T) {
                // [B,H,D,S] (d-major) so P@V B-fragments are key-contiguous
                int b = row >> 10, s = row & 1023, h = col >> 6, d = col & 63;
                ((bf16*)Cout)[(((size_t)b * NH + h) * HD + d) * SQ + s] = (bf16)val;
            } else if (MODE == EPI_RES) {
                ((float*)Cout)[idx] = val + resid[idx];
            } else { // exact GELU
                float g = 0.5f * val * (1.0f + erff(val * 0.70710678f));
                ((bf16*)Cout)[idx] = (bf16)g;
            }
        }
    }
}

// ---------------------------------------------------------------------------
// Flash attention. Q,K in [B,H,S,D]; V TRANSPOSED in [B,H,D,S].
// Block = 128 thr = 4 waves; each wave owns 16 query rows of one (b,h).
// Per 32-key block: 4 WMMA (QK^T over D=64) + online softmax (16-lane
// shfl_xor row reductions in C layout) + LDS transpose of P + 4 WMMA (P@V).
// All fragment loads are contiguous 16/32-byte vector loads.
// ---------------------------------------------------------------------------
__global__ __launch_bounds__(128)
void attention_kernel(const bf16* __restrict__ Q, const bf16* __restrict__ Km,
                      const bf16* __restrict__ Vt, const int* __restrict__ amask,
                      bf16* __restrict__ Out) {
    const int lane  = threadIdx.x & 31;
    const int wave  = threadIdx.x >> 5;   // 0..3
    const int half  = lane >> 4;
    const int nl    = lane & 15;
    const int bh    = blockIdx.y;         // 0..255
    const int batch = bh >> 4;
    const int head  = bh & 15;
    const int q0    = blockIdx.x * 64 + wave * 16;
    const size_t base = (size_t)bh * SQ * HD;   // same footprint for both layouts

    __shared__ bf16 pshare[4][16][32];

    // Q A-fragments for d-chunks [0,32) and [32,64)
    v16bf aq[2];
    {
        const bf16* Qrow = Q + base + (size_t)(q0 + nl) * HD;
#pragma unroll
        for (int c = 0; c < 2; ++c) {
            v8bf lo = *(const v8bf*)(Qrow + 32 * c + 16 * half);
            v8bf hi = *(const v8bf*)(Qrow + 32 * c + 16 * half + 8);
#pragma unroll
            for (int e = 0; e < 8; ++e) { aq[c][e] = lo[e]; aq[c][8 + e] = hi[e]; }
        }
    }

    float mrow[8], lrow[8];
#pragma unroll
    for (int v = 0; v < 8; ++v) { mrow[v] = -1.0e30f; lrow[v] = 0.0f; }
    v8f o[4] = {};

    const int* mb = amask + batch * SQ;

    for (int kb = 0; kb < SQ; kb += 32) {
        // ---- scores: S[16x32] as two 16x16 accumulators ----
        v8f s[2] = {};
#pragma unroll
        for (int j = 0; j < 2; ++j) {
#pragma unroll
            for (int c = 0; c < 2; ++c) {
                // B fragment: col = key (kb+j*16+nl), K = d = c*32 + 16*half + e
                const bf16* Kp = Km + base + (size_t)(kb + j * 16 + nl) * HD
                                 + 32 * c + 16 * half;
                v16bf b = *(const v16bf*)(Kp);
                s[j] = __builtin_amdgcn_wmma_f32_16x16x32_bf16(
                    false, aq[c], false, b, (short)0, s[j], false, false);
            }
        }

        // scale + mask (per-lane column)
        const bool ok0 = mb[kb + nl] != 0;
        const bool ok1 = mb[kb + 16 + nl] != 0;
#pragma unroll
        for (int v = 0; v < 8; ++v) {
            s[0][v] = ok0 ? s[0][v] * 0.125f : -3.0e38f;
            s[1][v] = ok1 ? s[1][v] * 0.125f : -3.0e38f;
        }

        // ---- online softmax per row (row v+8*half lives in one 16-lane group)
#pragma unroll
        for (int v = 0; v < 8; ++v) {
            float m = fmaxf(s[0][v], s[1][v]);
#pragma unroll
            for (int off = 1; off < 16; off <<= 1)
                m = fmaxf(m, __shfl_xor(m, off, 32));
            float mnew = fmaxf(mrow[v], m);
            float corr = __expf(mrow[v] - mnew);
            float p0 = __expf(s[0][v] - mnew);
            float p1 = __expf(s[1][v] - mnew);
            float rs = p0 + p1;
#pragma unroll
            for (int off = 1; off < 16; off <<= 1)
                rs += __shfl_xor(rs, off, 32);
            lrow[v] = lrow[v] * corr + rs;
            mrow[v] = mnew;
#pragma unroll
            for (int t = 0; t < 4; ++t) o[t][v] *= corr;
            int row = v + 8 * half;
            pshare[wave][row][nl]      = (bf16)p0;
            pshare[wave][row][16 + nl] = (bf16)p1;
        }
        asm volatile("s_wait_dscnt 0" ::: "memory");

        // ---- P as A-fragment (from LDS), V^T B-fragments (contiguous) ----
        v16bf ap;
#pragma unroll
        for (int e = 0; e < 16; ++e) {
            int kk = 16 * half + e;
            ap[e] = pshare[wave][nl][kk];
        }
#pragma unroll
        for (int t = 0; t < 4; ++t) {
            // B fragment: col = d (t*16+nl), K = key = kb + 16*half + e
            const bf16* Vp = Vt + base + (size_t)(t * 16 + nl) * SQ + kb + 16 * half;
            v16bf bv = *(const v16bf*)Vp;
            o[t] = __builtin_amdgcn_wmma_f32_16x16x32_bf16(
                false, ap, false, bv, (short)0, o[t], false, false);
        }
    }

    // normalize + scatter to [token, h*64+d] bf16
#pragma unroll
    for (int t = 0; t < 4; ++t) {
#pragma unroll
        for (int v = 0; v < 8; ++v) {
            int row = q0 + v + 8 * half;
            int tok = batch * SQ + row;
            float val = o[t][v] / lrow[v];
            Out[(size_t)tok * EMB_D + head * HD + t * 16 + nl] = (bf16)val;
        }
    }
}

// ---------------------------------------------------------------------------
// LayerNorm over EMB=1024, one token per 256-thread block, writes bf16
// ---------------------------------------------------------------------------
__global__ __launch_bounds__(256)
void layernorm_kernel(const float* __restrict__ X, const float* __restrict__ gamma,
                      const float* __restrict__ beta, bf16* __restrict__ Out) {
    __shared__ float red[2][8];
    int tok = blockIdx.x;
    const float* row = X + (size_t)tok * EMB_D;
    float vals[4], s = 0.f, ss = 0.f;
#pragma unroll
    for (int i = 0; i < 4; ++i) {
        float v = row[threadIdx.x + 256 * i];
        vals[i] = v; s += v; ss += v * v;
    }
#pragma unroll
    for (int off = 1; off < 32; off <<= 1) {
        s  += __shfl_xor(s, off, 32);
        ss += __shfl_xor(ss, off, 32);
    }
    if ((threadIdx.x & 31) == 0) {
        red[0][threadIdx.x >> 5] = s;
        red[1][threadIdx.x >> 5] = ss;
    }
    __syncthreads();
    s = 0.f; ss = 0.f;
#pragma unroll
    for (int i = 0; i < 8; ++i) { s += red[0][i]; ss += red[1][i]; }
    float mu  = s * (1.0f / EMB_D);
    float var = ss * (1.0f / EMB_D) - mu * mu;
    float rs  = rsqrtf(var + 1e-5f);
#pragma unroll
    for (int i = 0; i < 4; ++i) {
        int c = threadIdx.x + 256 * i;
        Out[(size_t)tok * EMB_D + c] = (bf16)((vals[i] - mu) * rs * gamma[c] + beta[c]);
    }
}

// ---------------------------------------------------------------------------
// Column-mean over the sequence axis: [B,S,EMB] -> [B,EMB]
// ---------------------------------------------------------------------------
__global__ void colmean_kernel(const float* __restrict__ X, float* __restrict__ Y) {
    int b = blockIdx.y;
    int c = blockIdx.x * 256 + threadIdx.x;
    const float* p = X + (size_t)b * SQ * EMB_D + c;
    float s = 0.f;
    for (int si = 0; si < SQ; ++si) s += p[(size_t)si * EMB_D];
    Y[b * EMB_D + c] = s * (1.0f / SQ);
}

// ---------------------------------------------------------------------------
// Final tiny projection: out[b,c] = Y[b,:] @ Wp[:,c] + bp[c]  (16x3)
// ---------------------------------------------------------------------------
__global__ void final_proj_kernel(const float* __restrict__ Y, const float* __restrict__ Wp,
                                  const float* __restrict__ bp, float* __restrict__ out) {
    int i = threadIdx.x;
    if (i < NB * 3) {
        int b = i / 3, c = i % 3;
        float s = 0.f;
        for (int k = 0; k < EMB_D; ++k) s += Y[b * EMB_D + k] * Wp[k * 3 + c];
        out[i] = s + bp[c];
    }
}

// ---------------------------------------------------------------------------
extern "C" void kernel_launch(void* const* d_in, const int* in_sizes, int n_in,
                              void* d_out, int out_size, void* d_ws, size_t ws_size,
                              hipStream_t stream) {
    const int*   ids   = (const int*)d_in[0];
    const int*   amask = (const int*)d_in[1];
    const float* emb   = (const float*)d_in[2];
    const float* Wq    = (const float*)d_in[3];
    const float* bq    = (const float*)d_in[4];
    const float* Wk    = (const float*)d_in[5];
    const float* bk    = (const float*)d_in[6];
    const float* Wv    = (const float*)d_in[7];
    const float* bv    = (const float*)d_in[8];
    const float* Wo    = (const float*)d_in[9];
    const float* bo    = (const float*)d_in[10];
    const float* W1    = (const float*)d_in[11];
    const float* b1    = (const float*)d_in[12];
    const float* W2    = (const float*)d_in[13];
    const float* b2    = (const float*)d_in[14];
    const float* Wp    = (const float*)d_in[15];
    const float* bp    = (const float*)d_in[16];
    const float* gamma = (const float*)d_in[17];
    const float* beta  = (const float*)d_in[18];

    char* ws = (char*)d_ws;
    size_t off = 0;
    auto alloc = [&](size_t bytes) -> void* {
        void* p = ws + off;
        off += (bytes + 255) & ~(size_t)255;
        return p;
    };

    float* x_f32 = (float*)alloc((size_t)TOK * EMB_D * 4);
    bf16*  x_b   = (bf16*) alloc((size_t)TOK * EMB_D * 2);
    bf16*  wq_t  = (bf16*) alloc((size_t)EMB_D * EMB_D * 2);  // [N,K] bf16
    bf16*  wk_t  = (bf16*) alloc((size_t)EMB_D * EMB_D * 2);
    bf16*  wv_t  = (bf16*) alloc((size_t)EMB_D * EMB_D * 2);
    bf16*  wo_t  = (bf16*) alloc((size_t)EMB_D * EMB_D * 2);
    bf16*  w1_t  = (bf16*) alloc((size_t)EMB_D * FF_D * 2);
    bf16*  w2_t  = (bf16*) alloc((size_t)FF_D * EMB_D * 2);
    bf16*  q_b   = (bf16*) alloc((size_t)TOK * EMB_D * 2);    // [B,H,S,D]
    bf16*  k_b   = (bf16*) alloc((size_t)TOK * EMB_D * 2);    // [B,H,S,D]
    bf16*  v_t   = (bf16*) alloc((size_t)TOK * EMB_D * 2);    // [B,H,D,S]
    bf16*  at_b  = (bf16*) alloc((size_t)TOK * EMB_D * 2);
    float* hpre  = (float*)alloc((size_t)TOK * EMB_D * 4);
    bf16*  h_b   = (bf16*) alloc((size_t)TOK * EMB_D * 2);
    bf16*  ff1_b = (bf16*) alloc((size_t)TOK * FF_D * 2);
    float* ff2_f = (float*)alloc((size_t)TOK * EMB_D * 4);
    float* ymean = (float*)alloc((size_t)NB * EMB_D * 4);

    // 1) weight fp32[K,N] -> bf16[N,K] transposed conversions
    cvt_transpose_kernel<<<dim3(EMB_D / 32, EMB_D / 32), 256, 0, stream>>>(Wq, wq_t, EMB_D, EMB_D);
    cvt_transpose_kernel<<<dim3(EMB_D / 32, EMB_D / 32), 256, 0, stream>>>(Wk, wk_t, EMB_D, EMB_D);
    cvt_transpose_kernel<<<dim3(EMB_D / 32, EMB_D / 32), 256, 0, stream>>>(Wv, wv_t, EMB_D, EMB_D);
    cvt_transpose_kernel<<<dim3(EMB_D / 32, EMB_D / 32), 256, 0, stream>>>(Wo, wo_t, EMB_D, EMB_D);
    cvt_transpose_kernel<<<dim3(FF_D / 32, EMB_D / 32), 256, 0, stream>>>(W1, w1_t, EMB_D, FF_D);
    cvt_transpose_kernel<<<dim3(EMB_D / 32, FF_D / 32), 256, 0, stream>>>(W2, w2_t, FF_D, EMB_D);

    // 2) embedding gather
    embed_gather_kernel<<<TOK, 256, 0, stream>>>(ids, emb, x_f32, x_b);

    // 3) QKV projections (WMMA); V scattered d-major for the P@V fragments
    dim3 gE(TOK / 16, EMB_D / 512);   // (1024, 2)
    gemm_bf16_kernel<EPI_QKV>  <<<gE, 256, 0, stream>>>(x_b, wq_t, bq, TOK, EMB_D, EMB_D, q_b, nullptr);
    gemm_bf16_kernel<EPI_QKV>  <<<gE, 256, 0, stream>>>(x_b, wk_t, bk, TOK, EMB_D, EMB_D, k_b, nullptr);
    gemm_bf16_kernel<EPI_QKV_T><<<gE, 256, 0, stream>>>(x_b, wv_t, bv, TOK, EMB_D, EMB_D, v_t, nullptr);

    // 4) flash attention (WMMA QK^T + WMMA PV, online softmax)
    attention_kernel<<<dim3(SQ / 64, NB * NH), 128, 0, stream>>>(q_b, k_b, v_t, amask, at_b);

    // 5) output projection + residual (fp32)
    gemm_bf16_kernel<EPI_RES><<<gE, 256, 0, stream>>>(at_b, wo_t, bo, TOK, EMB_D, EMB_D, hpre, x_f32);

    // 6) layernorm -> bf16
    layernorm_kernel<<<TOK, 256, 0, stream>>>(hpre, gamma, beta, h_b);

    // 7) FF1 + exact GELU -> bf16
    gemm_bf16_kernel<EPI_GELU><<<dim3(TOK / 16, FF_D / 512), 256, 0, stream>>>(
        h_b, w1_t, b1, TOK, FF_D, EMB_D, ff1_b, nullptr);

    // 8) FF2 -> fp32
    gemm_bf16_kernel<EPI_F32><<<gE, 256, 0, stream>>>(ff1_b, w2_t, b2, TOK, EMB_D, FF_D, ff2_f, nullptr);

    // 9) sequence mean + tiny projection
    colmean_kernel<<<dim3(EMB_D / 256, NB), 256, 0, stream>>>(ff2_f, ymean);
    final_proj_kernel<<<1, 64, 0, stream>>>(ymean, Wp, bp, (float*)d_out);
}
